// MambaBlock_63505386438822
// MI455X (gfx1250) — compile-verified, matrix operand order checked by reading, not run on hardware
//
#include <hip/hip_runtime.h>

// ---------------------------------------------------------------------------
// Types for CDNA5 WMMA / TDM
// ---------------------------------------------------------------------------
typedef __attribute__((ext_vector_type(16))) __bf16 v16bf;
typedef __attribute__((ext_vector_type(8)))  float  v8f;
typedef __attribute__((ext_vector_type(4)))  unsigned int v4u;
typedef __attribute__((ext_vector_type(8)))  int v8i;
typedef __attribute__((ext_vector_type(4)))  int v4i;

#if __has_builtin(__builtin_amdgcn_tensor_load_to_lds)
#define USE_TDM 1
#else
#define USE_TDM 0
#endif

// ---------------------------------------------------------------------------
// bf16 helpers (stored as raw ushort bit patterns in workspace)
// ---------------------------------------------------------------------------
__device__ __forceinline__ unsigned short f32_to_bf16(float f) {
    unsigned int x = __float_as_uint(f);
    unsigned int r = (x + 0x7FFFu + ((x >> 16) & 1u)) >> 16;
    return (unsigned short)r;
}
__device__ __forceinline__ float bf16_to_f32(unsigned short h) {
    return __uint_as_float(((unsigned int)h) << 16);
}

// Direction index mapping: position t in direction-d order -> original L index
// L = 4096 = 64*64, H = W = 64
__device__ __forceinline__ int dir_idx(int d, int t) {
    switch (d) {
        case 0:  return t;                               // lr
        case 1:  return 4095 - t;                        // rl
        case 2:  return (t & 63) * 64 + (t >> 6);        // tb (transpose HW)
        default: { int tt = 4095 - t;                    // bt
                   return (tt & 63) * 64 + (tt >> 6); }
    }
}

// ---------------------------------------------------------------------------
// TDM descriptor bit packing (CDNA5 ISA ch.8, D# groups, little-endian bits)
// ---------------------------------------------------------------------------
#if USE_TDM
__device__ __forceinline__ void desc_set(unsigned int* a, int lo, int len,
                                         unsigned long long v) {
    unsigned long long mask = (len >= 64) ? ~0ull : ((1ull << len) - 1ull);
    v &= mask;
    int d = lo >> 5, b = lo & 31;
    a[d] |= (unsigned int)(v << b);
    unsigned long long rest = (b == 0) ? (v >> 32) : (v >> (32 - b));
    if (b + len > 32) a[d + 1] |= (unsigned int)rest;
    if (b + len > 64) a[d + 2] |= (unsigned int)(rest >> 32);
}

// Issue a 2D bf16 tile load (tile_k x tile_rows) global -> LDS with row
// padding 16 DWORDs data + 4 DWORDs pad (i.e. LDS row stride 40 elements).
// rows_left: remaining valid rows for hardware OOB zero-fill.
__device__ __forceinline__ void tdm_load_tile(const unsigned short* gsrc,
                                              unsigned int lds_off,
                                              int K, int rows_left) {
    unsigned int g0a[4] = {0u, 0u, 0u, 0u};
    unsigned int g1a[8] = {0u, 0u, 0u, 0u, 0u, 0u, 0u, 0u};
    unsigned long long ga = (unsigned long long)(size_t)gsrc;

    desc_set(g0a, 0, 2, 1);                 // count = 1 (valid descriptor)
    desc_set(g0a, 32, 32, lds_off);         // lds_addr
    desc_set(g0a, 64, 57, ga);              // global_addr (tile start)
    desc_set(g0a, 126, 2, 2);               // type = 2 ("image")

    desc_set(g1a, 16, 2, 1);                // data_size = 1 -> 2 bytes
    desc_set(g1a, 20, 1, 1);                // pad_enable
    desc_set(g1a, 22, 3, 3);                // pad_interval: 16 DWORDs
    desc_set(g1a, 25, 7, 3);                // pad_amount: 4 DWORDs
    desc_set(g1a, 48, 32, (unsigned int)K);         // tensor_dim0
    desc_set(g1a, 80, 32, (unsigned int)rows_left); // tensor_dim1 (OOB clamp)
    desc_set(g1a, 112, 16, 32u);            // tile_dim0 = 32 (K step)
    desc_set(g1a, 128, 16, 128u);           // tile_dim1 = rows in tile
    desc_set(g1a, 160, 48, (unsigned int)K);        // tensor_dim0_stride

    v4u g0 = {g0a[0], g0a[1], g0a[2], g0a[3]};
    v8i g1 = {(int)g1a[0], (int)g1a[1], (int)g1a[2], (int)g1a[3],
              (int)g1a[4], (int)g1a[5], (int)g1a[6], (int)g1a[7]};
    v4i z4 = {0, 0, 0, 0};
    v8i z8 = {0, 0, 0, 0, 0, 0, 0, 0};
    // clang-23 / therock 6-arg form: (g0, g1, g2, g3, g_extra, cpol)
    __builtin_amdgcn_tensor_load_to_lds(g0, g1, z4, z4, z8, 0);
}
#endif

// ===========================================================================
// Generic elementwise f32 -> bf16 conversion
// ===========================================================================
__global__ void convert_bf16_kernel(const float* __restrict__ src,
                                    unsigned short* __restrict__ dst, int n) {
    int i = blockIdx.x * blockDim.x + threadIdx.x;
    if (i < n) dst[i] = f32_to_bf16(src[i]);
}

// ===========================================================================
// LayerNorm over D=256; one block per (b,l) row; outputs bf16 xn
// ===========================================================================
__global__ __launch_bounds__(256) void layernorm_kernel(
    const float* __restrict__ x, const float* __restrict__ gamma,
    const float* __restrict__ beta, unsigned short* __restrict__ xn_bf) {
    int row = blockIdx.x;
    int tid = threadIdx.x;
    float v = x[(size_t)row * 256 + tid];
    float s = v, sq = v * v;
    #pragma unroll
    for (int o = 16; o > 0; o >>= 1) {
        s  += __shfl_xor(s,  o, 32);
        sq += __shfl_xor(sq, o, 32);
    }
    __shared__ float ws1[8], ws2[8];
    if ((tid & 31) == 0) { ws1[tid >> 5] = s; ws2[tid >> 5] = sq; }
    __syncthreads();
    float tot = 0.f, totq = 0.f;
    #pragma unroll
    for (int i = 0; i < 8; ++i) { tot += ws1[i]; totq += ws2[i]; }
    float mu  = tot * (1.0f / 256.0f);
    float var = totq * (1.0f / 256.0f) - mu * mu;
    float xv  = (v - mu) * __frsqrt_rn(var + 1e-5f) * gamma[tid] + beta[tid];
    xn_bf[(size_t)row * 256 + tid] = f32_to_bf16(xv);
}

// ===========================================================================
// Tiled bf16 WMMA GEMM:  C[M,N] (+)= A[M,K] * W[N,K]^T
//   Block tile 128x64, 8 waves (wave32); each wave owns a 32x32 subtile:
//   2 A-frags x 2 B-frags -> 4x v_wmma_f32_16x16x32_bf16 per K-step of 32,
//   with operand-reuse hints between WMMAs sharing a fragment.
//   Tile staging: TDM tensor_load_to_lds (TENSORcnt) when available, with
//   hardware OOB zero-fill and LDS row padding; else cooperative b128 loads.
//   mode 0: C = acc    mode 1: C += acc
//   mode 2: C = addsrc + acc + bias[col]   (final projection epilogue)
// ===========================================================================
#define LDSW 40   // LDS row stride (elements): 32 data + 8 pad, 16B aligned

__global__ __launch_bounds__(256) void gemm_bf16_wmma(
    const unsigned short* __restrict__ A, const unsigned short* __restrict__ W,
    float* __restrict__ C, const float* __restrict__ addsrc,
    const float* __restrict__ bias, int M, int N, int K, int mode) {
    __shared__ unsigned short lA[128 * LDSW];
    __shared__ unsigned short lB[64 * LDSW];

    const int tid  = threadIdx.x;
    const int wave = tid >> 5;
    const int lane = tid & 31;
    const int wm   = wave >> 1;   // 0..3 -> M subtile of 32
    const int wn   = wave & 1;    // 0..1 -> N subtile of 32
    const int mbase = blockIdx.y * 128;
    const int nbase = blockIdx.x * 64;

    v8f acc00 = {}, acc01 = {}, acc10 = {}, acc11 = {};

    const int half = lane >> 4;
    const int l15  = lane & 15;
    const int mr0  = wm * 32 + l15;
    const int mr1  = mr0 + 16;
    const int nr0  = wn * 32 + l15;
    const int nr1  = nr0 + 16;

#if !USE_TDM
    const int lrow = tid >> 2;         // 0..63 cooperative-load row
    const int lcol = (tid & 3) * 8;    // 0,8,16,24 (8 bf16 = 16B per thread)
#endif

    for (int k0 = 0; k0 < K; k0 += 32) {
#if USE_TDM
        if (wave == 0) {
            // A tile: 128 rows x 32 K -> lA (DMA walks rows, zero-fills OOB)
            tdm_load_tile(A + (size_t)mbase * K + k0,
                          (unsigned int)(size_t)&lA[0], K, M - mbase);
            // W tile: 64 rows x 32 K -> lB (N<64 handled by OOB zero-fill)
            tdm_load_tile(W + (size_t)nbase * K + k0,
                          (unsigned int)(size_t)&lB[0], K,
                          (N - nbase) < 64 ? (N - nbase) : 64);
            __builtin_amdgcn_s_wait_tensorcnt(0);
        }
        __syncthreads();
#else
        {
            const unsigned short* src =
                A + (size_t)(mbase + lrow) * K + k0 + lcol;
            *(uint4*)&lA[lrow * LDSW + lcol] = *(const uint4*)src;
            *(uint4*)&lA[(lrow + 64) * LDSW + lcol] =
                *(const uint4*)(src + (size_t)64 * K);
        }
        {
            int n = nbase + lrow;
            uint4 v = make_uint4(0u, 0u, 0u, 0u);
            if (n < N) v = *(const uint4*)(W + (size_t)n * K + k0 + lcol);
            *(uint4*)&lB[lrow * LDSW + lcol] = v;
        }
        __syncthreads();
        if (k0 + 32 < K)
            __builtin_prefetch(A + (size_t)(mbase + lrow) * K + k0 + 32 + lcol,
                               0, 3);
#endif

        // --- fragment loads: 2x ds_load_b128 per fragment per lane ---
        union Frag { uint4 q[2]; v16bf v; } fa0, fa1, fb0, fb1;
        fa0.q[0] = *(const uint4*)&lA[mr0 * LDSW + half * 8];
        fa0.q[1] = *(const uint4*)&lA[mr0 * LDSW + 16 + half * 8];
        fa1.q[0] = *(const uint4*)&lA[mr1 * LDSW + half * 8];
        fa1.q[1] = *(const uint4*)&lA[mr1 * LDSW + 16 + half * 8];
        fb0.q[0] = *(const uint4*)&lB[nr0 * LDSW + half * 8];
        fb0.q[1] = *(const uint4*)&lB[nr0 * LDSW + 16 + half * 8];
        fb1.q[0] = *(const uint4*)&lB[nr1 * LDSW + half * 8];
        fb1.q[1] = *(const uint4*)&lB[nr1 * LDSW + 16 + half * 8];

        // 4 WMMAs; reuse hints on back-to-back ops sharing a fragment
        acc00 = __builtin_amdgcn_wmma_f32_16x16x32_bf16(
                    false, fa0.v, false, fb0.v, (short)0, acc00, false, false);
        acc01 = __builtin_amdgcn_wmma_f32_16x16x32_bf16(
                    false, fa0.v, false, fb1.v, (short)0, acc01, true, false);
        acc11 = __builtin_amdgcn_wmma_f32_16x16x32_bf16(
                    false, fa1.v, false, fb1.v, (short)0, acc11, false, true);
        acc10 = __builtin_amdgcn_wmma_f32_16x16x32_bf16(
                    false, fa1.v, false, fb0.v, (short)0, acc10, false, false);
        __syncthreads();
    }

    // --- epilogue: C/D layout -> lane = N col, VGPR j = M row (j | 8+j) ---
    const int c0 = nbase + wn * 32 + l15;
    const int c1 = c0 + 16;
    const int rb0 = mbase + wm * 32 + (half ? 8 : 0);
    const int rb1 = rb0 + 16;
    #pragma unroll
    for (int j = 0; j < 8; ++j) {
        int r0 = rb0 + j, r1 = rb1 + j;
        size_t i00 = (size_t)r0 * N + c0, i01 = (size_t)r0 * N + c1;
        size_t i10 = (size_t)r1 * N + c0, i11 = (size_t)r1 * N + c1;
        float v00 = acc00[j], v01 = acc01[j];
        float v10 = acc10[j], v11 = acc11[j];
        if (mode == 0) {
            if (c0 < N) { C[i00] = v00; C[i10] = v10; }
            if (c1 < N) { C[i01] = v01; C[i11] = v11; }
        } else if (mode == 1) {
            if (c0 < N) { C[i00] += v00; C[i10] += v10; }
            if (c1 < N) { C[i01] += v01; C[i11] += v11; }
        } else {
            if (c0 < N) {
                C[i00] = addsrc[i00] + v00 + bias[c0];
                C[i10] = addsrc[i10] + v10 + bias[c0];
            }
            if (c1 < N) {
                C[i01] = addsrc[i01] + v01 + bias[c1];
                C[i11] = addsrc[i11] + v11 + bias[c1];
            }
        }
    }
}

// ===========================================================================
// Causal depthwise conv1d (k=4) along direction-d order + bias + SiLU.
// ===========================================================================
__global__ __launch_bounds__(256) void conv_silu_kernel(
    const float* __restrict__ xz, const float* __restrict__ conv_w,
    const float* __restrict__ conv_b, unsigned short* __restrict__ xs_bf,
    int d) {
    int idx = blockIdx.x * blockDim.x + threadIdx.x;   // over B*L*512 = 2^22
    int c = idx & 511;
    int t = (idx >> 9) & 4095;
    int b = idx >> 21;
    const float* w = conv_w + ((size_t)d * 512 + c) * 4;
    float a = conv_b[d * 512 + c];
    #pragma unroll
    for (int k = 0; k < 4; ++k) {
        int tp = t - 3 + k;
        if (tp >= 0) {
            int l = dir_idx(d, tp);
            a += w[k] * xz[((size_t)b * 4096 + l) * 1024 + c];
        }
    }
    float sig = 1.0f / (1.0f + __expf(-a));
    xs_bf[(size_t)idx] = f32_to_bf16(a * sig);
}

// ===========================================================================
// delta = softplus(dt_r @ W_dt^T + b_dt); K=16 -> plain VALU FMAs
// ===========================================================================
__global__ __launch_bounds__(256) void delta_softplus_kernel(
    const float* __restrict__ xdbl, const float* __restrict__ W_dt,
    const float* __restrict__ b_dt, float* __restrict__ delta, int d) {
    int idx = blockIdx.x * blockDim.x + threadIdx.x;   // over B*L*512
    int c = idx & 511;
    int row = idx >> 9;
    const float* wr = W_dt + ((size_t)d * 512 + c) * 16;
    const float* xr = xdbl + (size_t)row * 48;
    float a = b_dt[d * 512 + c];
    #pragma unroll
    for (int r = 0; r < 16; ++r) a += wr[r] * xr[r];
    float sp = (a > 20.f) ? a : log1pf(__expf(a));
    delta[(size_t)idx] = sp;
}

// ===========================================================================
// Selective scan: sequential over t (4096), parallel over (b, c).
// 16 states per channel in regs; shared B/C via lane-strided load + readlane.
// ===========================================================================
__global__ __launch_bounds__(128) void ssm_scan_kernel(
    const float* __restrict__ delta, const unsigned short* __restrict__ xs_bf,
    const float* __restrict__ xdbl, const float* __restrict__ xz,
    const float* __restrict__ A_log, const float* __restrict__ D_ssm,
    unsigned short* __restrict__ y_bf, int d) {
    int b = blockIdx.x >> 2;
    int c = (blockIdx.x & 3) * 128 + threadIdx.x;
    int lane = threadIdx.x & 31;

    float Aarr[16];
    #pragma unroll
    for (int n = 0; n < 16; ++n)
        Aarr[n] = -__expf(A_log[((size_t)d * 512 + c) * 16 + n]);
    float Dv = D_ssm[d * 512 + c];

    float h[16];
    #pragma unroll
    for (int n = 0; n < 16; ++n) h[n] = 0.f;

    const size_t rowbase = (size_t)b * 4096;
    for (int t = 0; t < 4096; ++t) {
        size_t row = rowbase + t;
        float dt = delta[row * 512 + c];
        float u  = bf16_to_f32(xs_bf[row * 512 + c]);
        float bc = xdbl[row * 48 + 16 + lane];   // lanes: B[0..15], C[0..15]
        float du = dt * u;
        float y = 0.f;
        #pragma unroll
        for (int n = 0; n < 16; ++n) {
            float Bn = __shfl(bc, n, 32);
            float Cn = __shfl(bc, 16 + n, 32);
            h[n] = __expf(dt * Aarr[n]) * h[n] + du * Bn;
            y += h[n] * Cn;
        }
        int l = dir_idx(d, t);
        size_t orow = rowbase + l;
        float z = xz[orow * 1024 + 512 + c];
        float g = z / (1.0f + __expf(-z));
        y = (y + u * Dv) * g;
        y_bf[orow * 512 + c] = f32_to_bf16(y);
    }
}

// ===========================================================================
// Host-side launch sequence
// ===========================================================================
extern "C" void kernel_launch(void* const* d_in, const int* in_sizes, int n_in,
                              void* d_out, int out_size, void* d_ws, size_t ws_size,
                              hipStream_t stream) {
    (void)in_sizes; (void)n_in; (void)out_size; (void)ws_size;
    const float* x      = (const float*)d_in[0];
    const float* gamma  = (const float*)d_in[1];
    const float* beta   = (const float*)d_in[2];
    const float* W_in   = (const float*)d_in[3];
    const float* conv_w = (const float*)d_in[4];
    const float* conv_b = (const float*)d_in[5];
    const float* W_x    = (const float*)d_in[6];
    const float* W_dt   = (const float*)d_in[7];
    const float* b_dt   = (const float*)d_in[8];
    const float* A_log  = (const float*)d_in[9];
    const float* D_ssm  = (const float*)d_in[10];
    const float* W_mout = (const float*)d_in[11];
    const float* W_out  = (const float*)d_in[12];
    const float* b_out  = (const float*)d_in[13];
    float* out = (float*)d_out;

    const int B = 2, L = 4096, D = 256, DI = 512;
    const int M = B * L;                 // 8192 rows

    char* ws = (char*)d_ws;
    size_t off = 0;
    auto carve = [&](size_t bytes) -> char* {
        char* p = ws + off;
        off += (bytes + 255) & ~(size_t)255;
        return p;
    };
    unsigned short* xn_bf     = (unsigned short*)carve((size_t)M * D * 2);
    unsigned short* W_in_bf   = (unsigned short*)carve((size_t)4 * 1024 * 256 * 2);
    unsigned short* W_x_bf    = (unsigned short*)carve((size_t)4 * 48 * 512 * 2);
    unsigned short* W_mout_bf = (unsigned short*)carve((size_t)4 * 256 * 512 * 2);
    unsigned short* W_out_bf  = (unsigned short*)carve((size_t)256 * 256 * 2);
    float*          xz        = (float*)carve((size_t)M * 1024 * 4);
    unsigned short* xs_bf     = (unsigned short*)carve((size_t)M * DI * 2);
    float*          xdbl      = (float*)carve((size_t)M * 48 * 4);
    float*          dlt       = (float*)carve((size_t)M * DI * 4);
    unsigned short* y_bf      = (unsigned short*)carve((size_t)M * DI * 2);
    float*          merged    = (float*)carve((size_t)M * D * 4);
    unsigned short* merged_bf = (unsigned short*)carve((size_t)M * D * 2);

    auto cvt = [&](const float* src, unsigned short* dst, int n) {
        convert_bf16_kernel<<<(n + 255) / 256, 256, 0, stream>>>(src, dst, n);
    };

    // ---- 1: weight conversion to bf16 ----
    cvt(W_in,   W_in_bf,   4 * 1024 * 256);
    cvt(W_x,    W_x_bf,    4 * 48 * 512);
    cvt(W_mout, W_mout_bf, 4 * 256 * 512);
    cvt(W_out,  W_out_bf,  256 * 256);

    // ---- 2: LayerNorm -> xn_bf ----
    layernorm_kernel<<<M, 256, 0, stream>>>(x, gamma, beta, xn_bf);

    // ---- 3: per-direction pipeline ----
    for (int d = 0; d < 4; ++d) {
        gemm_bf16_wmma<<<dim3(1024 / 64, M / 128), 256, 0, stream>>>(
            xn_bf, W_in_bf + (size_t)d * 1024 * 256, xz,
            nullptr, nullptr, M, 1024, 256, 0);

        conv_silu_kernel<<<(B * L * DI) / 256, 256, 0, stream>>>(
            xz, conv_w, conv_b, xs_bf, d);

        gemm_bf16_wmma<<<dim3(1, M / 128), 256, 0, stream>>>(
            xs_bf, W_x_bf + (size_t)d * 48 * 512, xdbl,
            nullptr, nullptr, M, 48, 512, 0);

        delta_softplus_kernel<<<(B * L * DI) / 256, 256, 0, stream>>>(
            xdbl, W_dt, b_dt, dlt, d);

        ssm_scan_kernel<<<B * 4, 128, 0, stream>>>(
            dlt, xs_bf, xdbl, xz, A_log, D_ssm, y_bf, d);

        gemm_bf16_wmma<<<dim3(256 / 64, M / 128), 256, 0, stream>>>(
            y_bf, W_mout_bf + (size_t)d * 256 * 512, merged,
            nullptr, nullptr, M, 256, 512, (d == 0) ? 0 : 1);
    }

    // ---- 4: merged -> bf16 ----
    cvt(merged, merged_bf, M * D);

    // ---- 5: final projection + residual + bias -> d_out ----
    gemm_bf16_wmma<<<dim3(256 / 64, M / 128), 256, 0, stream>>>(
        merged_bf, W_out_bf, out, x, b_out, M, 256, 256, 2);
}
